// FunnelRelMultiheadAttention_14302241096297
// MI455X (gfx1250) — compile-verified, matrix operand order checked by reading, not run on hardware
//
#include <hip/hip_runtime.h>
#include <hip/hip_bf16.h>

// ---------------------------------------------------------------------------
// MI455X (gfx1250) Funnel relative multi-head attention, fused flash-style.
// All matmuls use v_wmma_f32_16x16x32_f16 (wave32, f16 in, f32 accumulate).
// GEMM waves compute 16x32 C-tiles (A-frag reuse); flash loop prefetches the
// next KV/pos tiles with global_prefetch_b8.
// ---------------------------------------------------------------------------

typedef __attribute__((ext_vector_type(16))) _Float16 v16h;
typedef __attribute__((ext_vector_type(8)))  float    v8f;

#define BATCH 4
#define SEQ   1024
#define DMODEL 768
#define NHEAD 12
#define HDIM  64
#define RLEN  2048
#define SCALE 0.125f           // 1/sqrt(64)
#define NEG_INF_PEN 1000000.0f

__device__ __forceinline__ v8f wmma_f16(v16h a, v16h b, v8f c) {
  // (neg_a, A, neg_b, B, c_mod, C, reuse_a, reuse_b)
  return __builtin_amdgcn_wmma_f32_16x16x32_f16(false, a, false, b, (short)0, c,
                                                false, false);
}

// ISA 7.12.2: 16-bit A 16x32 fragment. lane L: row = L%16, half = L/16.
// regs 0..3: K = half*8 + 2q {+0,+1}; regs 4..7: K = 16 + half*8 + 2(q-4).
__device__ __forceinline__ int frag_k(int q, int half) {
  return (q < 4) ? (half * 8 + q * 2) : (16 + half * 8 + (q - 4) * 2);
}

// A/B fragment where the per-lane fixed index (M or N) is the source ROW and
// K runs contiguously along the source row.  f16 source.
__device__ __forceinline__ v16h ld_frag_row_f16(const _Float16* src, int ld) {
  int lane = threadIdx.x & 31, r = lane & 15, half = lane >> 4;
  const _Float16* p = src + (size_t)r * ld;
  v16h f;
#pragma unroll
  for (int q = 0; q < 8; ++q) {
    int k = frag_k(q, half);
    f[2 * q]     = p[k];
    f[2 * q + 1] = p[k + 1];
  }
  return f;
}

// Same, f32 source with on-the-fly f16 convert.
__device__ __forceinline__ v16h ld_frag_row_f32(const float* src, int ld) {
  int lane = threadIdx.x & 31, r = lane & 15, half = lane >> 4;
  const float* p = src + (size_t)r * ld;
  v16h f;
#pragma unroll
  for (int q = 0; q < 8; ++q) {
    int k = frag_k(q, half);
    f[2 * q]     = (_Float16)p[k];
    f[2 * q + 1] = (_Float16)p[k + 1];
  }
  return f;
}

// B fragment where the per-lane fixed index is the source COLUMN and K runs
// down the rows (row-major [K][N] source, e.g. weight matrices). f32 source.
__device__ __forceinline__ v16h ld_frag_col_f32(const float* src, int ld) {
  int lane = threadIdx.x & 31, cidx = lane & 15, half = lane >> 4;
  const float* p = src + cidx;
  v16h f;
#pragma unroll
  for (int q = 0; q < 8; ++q) {
    int k = frag_k(q, half);
    f[2 * q]     = (_Float16)p[(size_t)k * ld];
    f[2 * q + 1] = (_Float16)p[(size_t)(k + 1) * ld];
  }
  return f;
}

// Same, f16 source (V matrix in P.V).
__device__ __forceinline__ v16h ld_frag_col_f16(const _Float16* src, int ld) {
  int lane = threadIdx.x & 31, cidx = lane & 15, half = lane >> 4;
  const _Float16* p = src + cidx;
  v16h f;
#pragma unroll
  for (int q = 0; q < 8; ++q) {
    int k = frag_k(q, half);
    f[2 * q]     = p[(size_t)k * ld];
    f[2 * q + 1] = p[(size_t)(k + 1) * ld];
  }
  return f;
}

// ---------------------------------------------------------------------------
// Kernel 1: QKV projections.  grid (256, 3, 3), block 256 (8 waves).
// Each wave computes a 16x32 C-tile (A-frag reused across two B tiles).
// z=0: qw/qr/qs = (query@Wq + r_*_bias)*scale ; z=1: k = key@Wk+bk ; z=2: v.
// ---------------------------------------------------------------------------
__global__ __launch_bounds__(256) void qkv_proj_kernel(
    const float* __restrict__ query, const float* __restrict__ key,
    const float* __restrict__ value, const float* __restrict__ Wq,
    const float* __restrict__ Wk, const float* __restrict__ Wv,
    const float* __restrict__ bk, const float* __restrict__ bv,
    const float* __restrict__ rwb, const float* __restrict__ rrb,
    const float* __restrict__ rsb, _Float16* __restrict__ qw,
    _Float16* __restrict__ qr, _Float16* __restrict__ qs,
    _Float16* __restrict__ kh, _Float16* __restrict__ vh) {
  const int z = blockIdx.z;
  const int w = threadIdx.x >> 5;
  const int n0 = (blockIdx.y * 8 + w) * 32;   // 32-wide per-wave tile
  const int m0 = blockIdx.x * 16;
  const int lane = threadIdx.x & 31, cc = lane & 15, half = lane >> 4;

  const float* X = (z == 0) ? query : ((z == 1) ? key : value);
  const float* W = (z == 0) ? Wq : ((z == 1) ? Wk : Wv);

  v8f acc0 = {}, acc1 = {};
  for (int kk = 0; kk < DMODEL; kk += 32) {
    v16h a  = ld_frag_row_f32(X + (size_t)m0 * DMODEL + kk, DMODEL);
    v16h b0 = ld_frag_col_f32(W + (size_t)kk * DMODEL + n0, DMODEL);
    v16h b1 = ld_frag_col_f32(W + (size_t)kk * DMODEL + n0 + 16, DMODEL);
    acc0 = wmma_f16(a, b0, acc0);
    acc1 = wmma_f16(a, b1, acc1);
  }

#pragma unroll
  for (int sub = 0; sub < 2; ++sub) {
    const v8f& acc = sub ? acc1 : acc0;
    const int ns = n0 + sub * 16;
    const int head = ns >> 6;
    const int h = (ns & 63) + cc;
#pragma unroll
    for (int p = 0; p < 8; ++p) {
      int i = m0 + p + 8 * half;
      int bb = i >> 10, s = i & 1023;
      size_t base = (((size_t)bb * NHEAD + head) * SEQ + s) * HDIM + h;
      float v = acc[p];
      if (z == 0) {
        qw[base] = (_Float16)((v + rwb[head * HDIM + h]) * SCALE);
        qr[base] = (_Float16)((v + rrb[head * HDIM + h]) * SCALE);
        qs[base] = (_Float16)((v + rsb[head * HDIM + h]) * SCALE);
      } else if (z == 1) {
        kh[base] = (_Float16)(v + bk[ns + cc]);
      } else {
        vh[base] = (_Float16)(v + bv[ns + cc]);
      }
    }
  }
}

// ---------------------------------------------------------------------------
// Kernel 2: r_head = pos_embed @ r_kernel  -> f16 [N][2048][64]
// grid (128, 3), block 256.  16x32 C-tile per wave.
// ---------------------------------------------------------------------------
__global__ __launch_bounds__(256) void rhead_kernel(
    const float* __restrict__ pos_embed, const float* __restrict__ r_kernel,
    _Float16* __restrict__ rh) {
  const int w = threadIdx.x >> 5;
  const int n0 = (blockIdx.y * 8 + w) * 32;
  const int m0 = blockIdx.x * 16;
  const int lane = threadIdx.x & 31, cc = lane & 15, half = lane >> 4;

  v8f acc0 = {}, acc1 = {};
  for (int kk = 0; kk < DMODEL; kk += 32) {
    v16h a  = ld_frag_row_f32(pos_embed + (size_t)m0 * DMODEL + kk, DMODEL);
    v16h b0 = ld_frag_col_f32(r_kernel + (size_t)kk * DMODEL + n0, DMODEL);
    v16h b1 = ld_frag_col_f32(r_kernel + (size_t)kk * DMODEL + n0 + 16, DMODEL);
    acc0 = wmma_f16(a, b0, acc0);
    acc1 = wmma_f16(a, b1, acc1);
  }
#pragma unroll
  for (int sub = 0; sub < 2; ++sub) {
    const v8f& acc = sub ? acc1 : acc0;
    const int ns = n0 + sub * 16;
    const int head = ns >> 6;
    const int h = (ns & 63) + cc;
#pragma unroll
    for (int p = 0; p < 8; ++p) {
      int t = m0 + p + 8 * half;
      rh[((size_t)head * RLEN + t) * HDIM + h] = (_Float16)acc[p];
    }
  }
}

// ---------------------------------------------------------------------------
// Kernel 3: segment bias dots  segb[b,n,s,t] = qs[b,n,s,:] . seg_embed[t,n,:]
// ---------------------------------------------------------------------------
__global__ __launch_bounds__(256) void seg_kernel(
    const _Float16* __restrict__ qs, const float* __restrict__ seg_embed,
    float* __restrict__ segb) {
  int tid = blockIdx.x * blockDim.x + threadIdx.x;
  if (tid >= BATCH * NHEAD * SEQ) return;
  int s = tid & 1023;
  int bn = tid >> 10;
  int n = bn % NHEAD;
  const _Float16* q = qs + ((size_t)bn * SEQ + s) * HDIM;
  const float* e0 = seg_embed + (size_t)n * HDIM;             // t=0
  const float* e1 = seg_embed + (size_t)(NHEAD + n) * HDIM;   // t=1
  float s0 = 0.f, s1 = 0.f;
#pragma unroll
  for (int h = 0; h < HDIM; ++h) {
    float qv = (float)q[h];
    s0 += qv * e0[h];
    s1 += qv * e1[h];
  }
  segb[(size_t)tid * 2 + 0] = s0;
  segb[(size_t)tid * 2 + 1] = s1;
}

// ---------------------------------------------------------------------------
// Kernel 4: fused flash attention.  grid (S/16, B*N), block 32 (one wave).
// Per 32-col KV step: 4 WMMA content + 6 WMMA banded pos + 4 WMMA P.V.
// rel_shift: pos[i,j] = QR[i] . RH[1024 + j - i]; band fits 3 t-tiles.
// Next-step K/V/RH tiles are prefetched (global_prefetch_b8).
// ---------------------------------------------------------------------------
__global__ __launch_bounds__(32) void flash_attn_kernel(
    const _Float16* __restrict__ qw, const _Float16* __restrict__ qr,
    const _Float16* __restrict__ kh, const _Float16* __restrict__ vh,
    const _Float16* __restrict__ rh, const float* __restrict__ segb,
    const unsigned char* __restrict__ ttm, const float* __restrict__ amask,
    const float* __restrict__ clsm, _Float16* __restrict__ attn) {
  __shared__ float    psh[16][49];   // 48-wide positional band (+pad)
  __shared__ _Float16 pbuf[16][32];  // P tile staging for D->A transpose

  const int bn = blockIdx.y;
  const int b = bn / NHEAD, n = bn % NHEAD;
  const int i0 = blockIdx.x * 16;
  const int lane = threadIdx.x & 31, cc = lane & 15, half = lane >> 4;

  const size_t qbase = ((size_t)bn * SEQ + i0) * HDIM;
  v16h aqw0 = ld_frag_row_f16(qw + qbase, HDIM);
  v16h aqw1 = ld_frag_row_f16(qw + qbase + 32, HDIM);
  v16h aqr0 = ld_frag_row_f16(qr + qbase, HDIM);
  v16h aqr1 = ld_frag_row_f16(qr + qbase + 32, HDIM);

  float mi[8], li[8], sg0[8], sg1[8];
  v8f o0 = {}, o1 = {}, o2 = {}, o3 = {};
#pragma unroll
  for (int p = 0; p < 8; ++p) {
    mi[p] = -3.0e38f;
    li[p] = 0.f;
    int i = i0 + p + 8 * half;
    sg0[p] = segb[((size_t)bn * SEQ + i) * 2 + 0];
    sg1[p] = segb[((size_t)bn * SEQ + i) * 2 + 1];
  }

  for (int j0 = 0; j0 < SEQ; j0 += 32) {
    // ---- prefetch next step's K/V rows and RH band into caches ----
    if (j0 + 32 < SEQ) {
      size_t kbn = ((size_t)bn * SEQ + j0 + 32) * HDIM;
      size_t rbn = ((size_t)n * RLEN + (SEQ + j0 + 32 - i0 - 16)) * HDIM;
      // one lane-spread byte per 128B row; 32 lanes cover the tiles
      __builtin_prefetch(kh + kbn + lane * HDIM, 0, 3);
      __builtin_prefetch(vh + kbn + lane * HDIM, 0, 3);
      __builtin_prefetch(rh + rbn + lane * HDIM, 0, 3);
      __builtin_prefetch(rh + rbn + (32 + lane) * HDIM, 0, 3);
    }

    // ---- content scores: Qw (16x64) x K^T, two 16-col subtiles ----
    v8f s0 = {}, s1 = {};
    {
      size_t kb = ((size_t)bn * SEQ + j0) * HDIM;
      v16h b00 = ld_frag_row_f16(kh + kb, HDIM);
      v16h b10 = ld_frag_row_f16(kh + kb + 16 * HDIM, HDIM);
      s0 = wmma_f16(aqw0, b00, s0);
      s1 = wmma_f16(aqw0, b10, s1);
      v16h b01 = ld_frag_row_f16(kh + kb + 32, HDIM);
      v16h b11 = ld_frag_row_f16(kh + kb + 16 * HDIM + 32, HDIM);
      s0 = wmma_f16(aqw1, b01, s0);
      s1 = wmma_f16(aqw1, b11, s1);
    }
    // ---- positional band: Qr x RH^T over t in [t0-16, t0+32) ----
    const int t0 = SEQ + j0 - i0;
    v8f p0 = {}, p1 = {}, p2 = {};
    {
      size_t rb = ((size_t)n * RLEN + (t0 - 16)) * HDIM;
      v16h r0 = ld_frag_row_f16(rh + rb, HDIM);
      v16h r1 = ld_frag_row_f16(rh + rb + 16 * HDIM, HDIM);
      v16h r2 = ld_frag_row_f16(rh + rb + 32 * HDIM, HDIM);
      p0 = wmma_f16(aqr0, r0, p0);
      p1 = wmma_f16(aqr0, r1, p1);
      p2 = wmma_f16(aqr0, r2, p2);
      r0 = ld_frag_row_f16(rh + rb + 32, HDIM);
      r1 = ld_frag_row_f16(rh + rb + 16 * HDIM + 32, HDIM);
      r2 = ld_frag_row_f16(rh + rb + 32 * HDIM + 32, HDIM);
      p0 = wmma_f16(aqr1, r0, p0);
      p1 = wmma_f16(aqr1, r1, p1);
      p2 = wmma_f16(aqr1, r2, p2);
    }
#pragma unroll
    for (int p = 0; p < 8; ++p) {
      int m = p + 8 * half;
      psh[m][cc] = p0[p];
      psh[m][16 + cc] = p1[p];
      psh[m][32 + cc] = p2[p];
    }
    __syncthreads();

    // ---- assemble scores: content + shifted pos + token-type + mask ----
    const float pen0 = -NEG_INF_PEN * (1.0f - amask[b * SEQ + j0 + cc]);
    const float pen1 = -NEG_INF_PEN * (1.0f - amask[b * SEQ + j0 + 16 + cc]);
#pragma unroll
    for (int p = 0; p < 8; ++p) {
      int m = p + 8 * half, i = i0 + m;
      {
        int j = j0 + cc;
        float cm = clsm[(size_t)i * SEQ + j];
        float pos = psh[m][cc - m + 16] * cm;   // t = 1024 + j - i
        float tt = (ttm[((size_t)b * SEQ + i) * SEQ + j] ? sg1[p] : sg0[p]) * cm;
        s0[p] = s0[p] + pos + tt + pen0;
      }
      {
        int j = j0 + 16 + cc;
        float cm = clsm[(size_t)i * SEQ + j];
        float pos = psh[m][cc - m + 32] * cm;
        float tt = (ttm[((size_t)b * SEQ + i) * SEQ + j] ? sg1[p] : sg0[p]) * cm;
        s1[p] = s1[p] + pos + tt + pen1;
      }
    }
    __syncthreads();

    // ---- online softmax (reductions stay inside 16-lane C-frag groups) ----
#pragma unroll
    for (int p = 0; p < 8; ++p) {
      float vmax = fmaxf(s0[p], s1[p]);
      vmax = fmaxf(vmax, __shfl_xor(vmax, 1, 32));
      vmax = fmaxf(vmax, __shfl_xor(vmax, 2, 32));
      vmax = fmaxf(vmax, __shfl_xor(vmax, 4, 32));
      vmax = fmaxf(vmax, __shfl_xor(vmax, 8, 32));
      float mnew = fmaxf(mi[p], vmax);
      float corr = __expf(mi[p] - mnew);
      float e0 = __expf(s0[p] - mnew);
      float e1 = __expf(s1[p] - mnew);
      float rs = e0 + e1;
      rs += __shfl_xor(rs, 1, 32);
      rs += __shfl_xor(rs, 2, 32);
      rs += __shfl_xor(rs, 4, 32);
      rs += __shfl_xor(rs, 8, 32);
      li[p] = li[p] * corr + rs;
      mi[p] = mnew;
      o0[p] *= corr; o1[p] *= corr; o2[p] *= corr; o3[p] *= corr;
      int m = p + 8 * half;
      pbuf[m][cc] = (_Float16)e0;
      pbuf[m][16 + cc] = (_Float16)e1;
    }
    __syncthreads();

    // ---- O += P (16x32) x V (32x64) ----
    v16h ap = ld_frag_row_f16(&pbuf[0][0], 32);
    size_t vb = ((size_t)bn * SEQ + j0) * HDIM;
    v16h vf0 = ld_frag_col_f16(vh + vb + 0, HDIM);
    v16h vf1 = ld_frag_col_f16(vh + vb + 16, HDIM);
    v16h vf2 = ld_frag_col_f16(vh + vb + 32, HDIM);
    v16h vf3 = ld_frag_col_f16(vh + vb + 48, HDIM);
    o0 = wmma_f16(ap, vf0, o0);
    o1 = wmma_f16(ap, vf1, o1);
    o2 = wmma_f16(ap, vf2, o2);
    o3 = wmma_f16(ap, vf3, o3);
    __syncthreads();
  }

  // ---- epilogue: normalize and store attn_vec f16 [B*S][N*H] ----
#pragma unroll
  for (int p = 0; p < 8; ++p) {
    int i = i0 + p + 8 * half;
    float inv = 1.0f / li[p];
    size_t ob = ((size_t)b * SEQ + i) * DMODEL + n * HDIM;
    attn[ob + 0 + cc]  = (_Float16)(o0[p] * inv);
    attn[ob + 16 + cc] = (_Float16)(o1[p] * inv);
    attn[ob + 32 + cc] = (_Float16)(o2[p] * inv);
    attn[ob + 48 + cc] = (_Float16)(o3[p] * inv);
  }
}

// ---------------------------------------------------------------------------
// Kernel 5: out = attn_vec @ Wo + bo + query (residual) -> f32 x buffer
// grid (256, 3), block 256.  16x32 C-tile per wave.
// ---------------------------------------------------------------------------
__global__ __launch_bounds__(256) void outproj_kernel(
    const _Float16* __restrict__ attn, const float* __restrict__ Wo,
    const float* __restrict__ bo, const float* __restrict__ query,
    float* __restrict__ x) {
  const int w = threadIdx.x >> 5;
  const int n0 = (blockIdx.y * 8 + w) * 32;
  const int m0 = blockIdx.x * 16;
  const int lane = threadIdx.x & 31, cc = lane & 15, half = lane >> 4;

  v8f acc0 = {}, acc1 = {};
  for (int kk = 0; kk < DMODEL; kk += 32) {
    v16h a  = ld_frag_row_f16(attn + (size_t)m0 * DMODEL + kk, DMODEL);
    v16h b0 = ld_frag_col_f32(Wo + (size_t)kk * DMODEL + n0, DMODEL);
    v16h b1 = ld_frag_col_f32(Wo + (size_t)kk * DMODEL + n0 + 16, DMODEL);
    acc0 = wmma_f16(a, b0, acc0);
    acc1 = wmma_f16(a, b1, acc1);
  }
#pragma unroll
  for (int sub = 0; sub < 2; ++sub) {
    const v8f& acc = sub ? acc1 : acc0;
    const int ns = n0 + sub * 16;
#pragma unroll
    for (int p = 0; p < 8; ++p) {
      int i = m0 + p + 8 * half;
      size_t idx = (size_t)i * DMODEL + ns + cc;
      x[idx] = acc[p] + bo[ns + cc] + query[idx];
    }
  }
}

// ---------------------------------------------------------------------------
// Kernel 6: row LayerNorm over 768.  grid 4096, block 256 (3 elems/thread).
// ---------------------------------------------------------------------------
__global__ __launch_bounds__(256) void layernorm_kernel(
    const float* __restrict__ x, const float* __restrict__ g,
    const float* __restrict__ bb, float* __restrict__ out) {
  __shared__ float red[256];
  const int row = blockIdx.x;
  const int t = threadIdx.x;
  const float* xr = x + (size_t)row * DMODEL;
  float v0 = xr[t], v1 = xr[t + 256], v2 = xr[t + 512];
  red[t] = v0 + v1 + v2;
  __syncthreads();
  for (int off = 128; off > 0; off >>= 1) {
    if (t < off) red[t] += red[t + off];
    __syncthreads();
  }
  float mu = red[0] * (1.0f / DMODEL);
  __syncthreads();
  float d0 = v0 - mu, d1 = v1 - mu, d2 = v2 - mu;
  red[t] = d0 * d0 + d1 * d1 + d2 * d2;
  __syncthreads();
  for (int off = 128; off > 0; off >>= 1) {
    if (t < off) red[t] += red[t + off];
    __syncthreads();
  }
  float rstd = rsqrtf(red[0] * (1.0f / DMODEL) + 1e-9f);
  float* orow = out + (size_t)row * DMODEL;
  orow[t]       = d0 * rstd * g[t]       + bb[t];
  orow[t + 256] = d1 * rstd * g[t + 256] + bb[t + 256];
  orow[t + 512] = d2 * rstd * g[t + 512] + bb[t + 512];
}

// ---------------------------------------------------------------------------
extern "C" void kernel_launch(void* const* d_in, const int* in_sizes, int n_in,
                              void* d_out, int out_size, void* d_ws,
                              size_t ws_size, hipStream_t stream) {
  const float* query     = (const float*)d_in[0];
  const float* key       = (const float*)d_in[1];
  const float* value     = (const float*)d_in[2];
  const float* pos_embed = (const float*)d_in[3];
  const unsigned char* ttm = (const unsigned char*)d_in[4];
  const float* amask     = (const float*)d_in[5];
  const float* clsm      = (const float*)d_in[6];
  const float* Wq        = (const float*)d_in[7];
  const float* Wk        = (const float*)d_in[8];
  const float* bk        = (const float*)d_in[9];
  const float* Wv        = (const float*)d_in[10];
  const float* bv        = (const float*)d_in[11];
  const float* rwb       = (const float*)d_in[12];
  const float* rrb       = (const float*)d_in[13];
  const float* r_kernel  = (const float*)d_in[14];
  const float* rsb       = (const float*)d_in[15];
  const float* seg_embed = (const float*)d_in[16];
  const float* Wo        = (const float*)d_in[17];
  const float* bo        = (const float*)d_in[18];
  const float* ln_g      = (const float*)d_in[19];
  const float* ln_b      = (const float*)d_in[20];
  float* out = (float*)d_out;

  // workspace layout
  const size_t QS = (size_t)BATCH * NHEAD * SEQ * HDIM;  // per-head f16 tensor
  _Float16* qw = (_Float16*)d_ws;
  _Float16* qr = qw + QS;
  _Float16* qs = qr + QS;
  _Float16* kh = qs + QS;
  _Float16* vh = kh + QS;
  _Float16* rh = vh + QS;                                   // [N][2048][64]
  float* segb = (float*)(rh + (size_t)NHEAD * RLEN * HDIM); // [B][N][S][2]
  _Float16* attn = (_Float16*)(segb + (size_t)BATCH * NHEAD * SEQ * 2);
  float* xbuf = (float*)(attn + (size_t)BATCH * SEQ * DMODEL);

  qkv_proj_kernel<<<dim3(256, 3, 3), 256, 0, stream>>>(
      query, key, value, Wq, Wk, Wv, bk, bv, rwb, rrb, rsb,
      qw, qr, qs, kh, vh);

  rhead_kernel<<<dim3(128, 3), 256, 0, stream>>>(pos_embed, r_kernel, rh);

  seg_kernel<<<dim3((BATCH * NHEAD * SEQ + 255) / 256), 256, 0, stream>>>(
      qs, seg_embed, segb);

  flash_attn_kernel<<<dim3(SEQ / 16, BATCH * NHEAD), 32, 0, stream>>>(
      qw, qr, kh, vh, rh, segb, ttm, amask, clsm, attn);

  outproj_kernel<<<dim3(256, 3), 256, 0, stream>>>(attn, Wo, bo, query, xbuf);

  layernorm_kernel<<<dim3(BATCH * SEQ), 256, 0, stream>>>(xbuf, ln_g, ln_b, out);
}
